// SparseBasicBlock_15504831938850
// MI455X (gfx1250) — matrix operand order repaired and checked by reading, not compiled, not run on hardware
//
#include <hip/hip_runtime.h>

// ---------------- CDNA5 WMMA types ----------------
typedef __attribute__((ext_vector_type(16))) __bf16 v16bf;
typedef __attribute__((ext_vector_type(8)))  float  v8f;

#define NVOX  400000
#define CCH   128
#define KOFF  9
#define MTILE 64                 // rows per block
#define NTILES 4                 // 16-row M-tiles per wave
#define EPSBN 1e-4f
#define NPACK (KOFF*4*8*32*16)   // 147456 packed bf16 weight elements per conv
#define NRED  625                // column-reduction blocks (625*640 = 400000)
#define RPB   640

union FragBF { v16bf v; uint4 q[2]; };

static __device__ __forceinline__ unsigned short f2bf(float f) {
  // round-to-nearest-even fp32 -> bf16
  unsigned u = __float_as_uint(f);
  u += 0x7fffu + ((u >> 16) & 1u);
  return (unsigned short)(u >> 16);
}

// ---------------- prep: fp32 -> bf16 activations ----------------
__global__ __launch_bounds__(256) void cvt_bf16_k(const float* __restrict__ x,
                                                  unsigned short* __restrict__ o,
                                                  int n) {
  for (int i = blockIdx.x * blockDim.x + threadIdx.x; i < n; i += gridDim.x * blockDim.x)
    o[i] = f2bf(x[i]);
}

// ---------------- prep: pack W[k][cin][cout] into per-lane B fragments ----
// dst layout: frag f = (k*4+kk)*8 + ncg ; within frag: lane*16 + e (bf16)
// lane: n = lane&15 (cout within group), h = lane>>4 ; element e -> K = kk*32 + 16h + e
__global__ __launch_bounds__(256) void pack_w_k(const float* __restrict__ W,
                                                unsigned short* __restrict__ wp) {
  int t = blockIdx.x * blockDim.x + threadIdx.x;
  if (t >= NPACK) return;
  int e    = t & 15;
  int lane = (t >> 4) & 31;
  int ncg  = (t >> 9) & 7;
  int kk   = (t >> 12) & 3;
  int k    = t >> 14;
  int h = lane >> 4, n = lane & 15;
  int K    = kk * 32 + h * 16 + e;     // cin within the 128-deep slice
  int cout = ncg * 16 + n;
  wp[t] = f2bf(W[k * CCH * CCH + K * CCH + cout]);
}

// ---------------- the heavy kernel: rulebook gather + WMMA GEMM ----------
// One block = 64 voxels, 8 waves; wave w computes cout columns [16w,16w+16)
// for 4 independent 16-row M-tiles (each B fragment feeds 4 WMMAs).
// A tile (64 x 128 bf16) staged in LDS per kernel offset k, double buffered;
// global gather for k+1 overlaps WMMA on k.
__global__ __launch_bounds__(256) void conv_wmma_k(const uint4* __restrict__ xq,  // N x 16 uint4 bf16 rows
                                                   const uint4* __restrict__ wq,  // packed weights
                                                   float* __restrict__ y,         // N x 128 fp32 out
                                                   const int* __restrict__ nbr_idx,
                                                   const unsigned char* __restrict__ nbr_mask) {
  __shared__ int  s_idx[MTILE * KOFF];     // 576 ints
  __shared__ uint4 s_a[2][MTILE * 16];     // 2 x (64 rows x 256B) = 32 KB

  const int t  = threadIdx.x;
  const int m0 = blockIdx.x * MTILE;

  for (int i = t; i < MTILE * KOFF; i += 256) {   // coalesced rulebook load
    int idx = nbr_idx[m0 * KOFF + i];
    s_idx[i] = nbr_mask[m0 * KOFF + i] ? idx : -1;
  }
  __syncthreads();

  // stage k = 0 : 64 rows x 16 chunks = 1024 units, 4 per thread
#pragma unroll
  for (int u = 0; u < 4; ++u) {
    int uu = t + u * 256;
    int r = uu >> 4, ch = uu & 15;
    int idx = s_idx[r * KOFF];
    uint4 v = make_uint4(0u, 0u, 0u, 0u);
    if (idx >= 0) v = xq[idx * 16 + ch];
    s_a[0][uu] = v;
  }
  __syncthreads();

  const int lane = t & 31;
  const int wave = t >> 5;              // cout group 0..7
  const int h = lane >> 4;
  const int n = lane & 15;

  v8f acc[NTILES];
#pragma unroll
  for (int mt = 0; mt < NTILES; ++mt) acc[mt] = (v8f){};

  for (int k = 0; k < KOFF; ++k) {
    // prefetch gather for k+1 (overlaps with WMMA below)
    uint4 nv[4];
    const bool more = (k + 1 < KOFF);
    if (more) {
#pragma unroll
      for (int u = 0; u < 4; ++u) {
        int uu = t + u * 256;
        int r = uu >> 4, ch = uu & 15;
        int idx = s_idx[r * KOFF + k + 1];
        nv[u] = make_uint4(0u, 0u, 0u, 0u);
        if (idx >= 0) nv[u] = xq[idx * 16 + ch];
      }
    }
    const uint4* abuf = s_a[k & 1];
#pragma unroll
    for (int kk = 0; kk < 4; ++kk) {
      FragBF b;                          // one B fragment, reused by 4 M-tiles
      int wb = (((k * 4 + kk) * 8 + wave) * 64) + lane * 2;
      b.q[0] = wq[wb];
      b.q[1] = wq[wb + 1];
#pragma unroll
      for (int mt = 0; mt < NTILES; ++mt) {
        FragBF a;                        // A frag: row = mt*16 + n, K runs kk*32+16h..+15
        int ab = (mt * 16 + n) * 16 + kk * 4 + 2 * h;
        a.q[0] = abuf[ab];
        a.q[1] = abuf[ab + 1];
        acc[mt] = __builtin_amdgcn_wmma_f32_16x16x32_bf16(false, a.v, false, b.v,
                                                          (short)0, acc[mt], false, false);
      }
    }
    if (more) {
#pragma unroll
      for (int u = 0; u < 4; ++u) s_a[(k + 1) & 1][t + u * 256] = nv[u];
    }
    __syncthreads();
  }

  // C/D layout: lane (n, h), VGPR v -> M = mt*16 + 8h + v, N = wave*16 + n
#pragma unroll
  for (int mt = 0; mt < NTILES; ++mt)
#pragma unroll
    for (int v = 0; v < 8; ++v)
      y[(m0 + mt * 16 + h * 8 + v) * CCH + wave * 16 + n] = acc[mt][v];
}

// ---------------- BN: per-block column partial sums (deterministic) ------
__global__ __launch_bounds__(CCH) void col_partial_k(const float* __restrict__ y,
                                                     float* __restrict__ ps,
                                                     float* __restrict__ pq) {
  int c = threadIdx.x;
  int r0 = blockIdx.x * RPB;
  int r1 = r0 + RPB; if (r1 > NVOX) r1 = NVOX;
  float s = 0.f, s2 = 0.f;
  for (int r = r0; r < r1; ++r) {
    float v = y[r * CCH + c];
    s += v; s2 += v * v;
  }
  ps[blockIdx.x * CCH + c] = s;
  pq[blockIdx.x * CCH + c] = s2;
}

__global__ __launch_bounds__(CCH) void bn_finalize_k(const float* __restrict__ ps,
                                                     const float* __restrict__ pq,
                                                     const float* __restrict__ gamma,
                                                     const float* __restrict__ beta,
                                                     float* __restrict__ scale,
                                                     float* __restrict__ shift) {
  int c = threadIdx.x;
  float s = 0.f, s2 = 0.f;
  for (int b = 0; b < NRED; ++b) { s += ps[b * CCH + c]; s2 += pq[b * CCH + c]; }
  float mu  = s  / (float)NVOX;
  float var = s2 / (float)NVOX - mu * mu;
  float sc  = gamma[c] * rsqrtf(var + EPSBN);
  scale[c] = sc;
  shift[c] = beta[c] - mu * sc;
}

// ---------------- BN+ReLU -> bf16 (conv2 input) --------------------------
__global__ __launch_bounds__(256) void bn_relu_bf16_k(const float* __restrict__ y,
                                                      const float* __restrict__ scale,
                                                      const float* __restrict__ shift,
                                                      unsigned short* __restrict__ xb,
                                                      int nelem) {
  for (int i = blockIdx.x * blockDim.x + threadIdx.x; i < nelem; i += gridDim.x * blockDim.x) {
    int c = i & (CCH - 1);
    float v = fmaxf(y[i] * scale[c] + shift[c], 0.f);
    xb[i] = f2bf(v);
  }
}

// ---------------- BN2 + residual + ReLU -> fp32 out ----------------------
__global__ __launch_bounds__(256) void bn_add_relu_k(const float* __restrict__ y,
                                                     const float* __restrict__ scale,
                                                     const float* __restrict__ shift,
                                                     const float* __restrict__ feat,
                                                     float* __restrict__ out,
                                                     int nelem) {
  for (int i = blockIdx.x * blockDim.x + threadIdx.x; i < nelem; i += gridDim.x * blockDim.x) {
    int c = i & (CCH - 1);
    out[i] = fmaxf(y[i] * scale[c] + shift[c] + feat[i], 0.f);
  }
}

// ---------------- launcher ----------------
extern "C" void kernel_launch(void* const* d_in, const int* in_sizes, int n_in,
                              void* d_out, int out_size, void* d_ws, size_t ws_size,
                              hipStream_t stream) {
  const float* feat   = (const float*)d_in[0];
  const float* W1     = (const float*)d_in[1];
  const float* W2     = (const float*)d_in[2];
  const float* gamma1 = (const float*)d_in[3];
  const float* beta1  = (const float*)d_in[4];
  const float* gamma2 = (const float*)d_in[5];
  const float* beta2  = (const float*)d_in[6];
  const int*   nbr_idx  = (const int*)d_in[7];
  const unsigned char* nbr_mask = (const unsigned char*)d_in[8];
  float* out = (float*)d_out;

  // workspace carve-out (256B aligned slabs)
  char* ws = (char*)d_ws;
  auto carve = [&](size_t bytes) { char* p = ws; ws += ((bytes + 255) & ~(size_t)255); return p; };
  unsigned short* xbf = (unsigned short*)carve((size_t)NVOX * CCH * 2);  // bf16 activations (reused)
  float* y            = (float*)carve((size_t)NVOX * CCH * 4);           // pre-BN conv out (reused)
  unsigned short* wp1 = (unsigned short*)carve((size_t)NPACK * 2);
  unsigned short* wp2 = (unsigned short*)carve((size_t)NPACK * 2);
  float* ps     = (float*)carve((size_t)NRED * CCH * 4);
  float* pq     = (float*)carve((size_t)NRED * CCH * 4);
  float* scale1 = (float*)carve(CCH * 4);
  float* shift1 = (float*)carve(CCH * 4);
  float* scale2 = (float*)carve(CCH * 4);
  float* shift2 = (float*)carve(CCH * 4);

  const int nelem = NVOX * CCH;
  const int egrid = 2048;

  cvt_bf16_k<<<egrid, 256, 0, stream>>>(feat, xbf, nelem);
  pack_w_k<<<(NPACK + 255) / 256, 256, 0, stream>>>(W1, wp1);
  pack_w_k<<<(NPACK + 255) / 256, 256, 0, stream>>>(W2, wp2);

  // conv1
  conv_wmma_k<<<NVOX / MTILE, 256, 0, stream>>>((const uint4*)xbf, (const uint4*)wp1, y,
                                                nbr_idx, nbr_mask);
  col_partial_k<<<NRED, CCH, 0, stream>>>(y, ps, pq);
  bn_finalize_k<<<1, CCH, 0, stream>>>(ps, pq, gamma1, beta1, scale1, shift1);
  bn_relu_bf16_k<<<egrid, 256, 0, stream>>>(y, scale1, shift1, xbf, nelem);

  // conv2
  conv_wmma_k<<<NVOX / MTILE, 256, 0, stream>>>((const uint4*)xbf, (const uint4*)wp2, y,
                                                nbr_idx, nbr_mask);
  col_partial_k<<<NRED, CCH, 0, stream>>>(y, ps, pq);
  bn_finalize_k<<<1, CCH, 0, stream>>>(ps, pq, gamma2, beta2, scale2, shift2);

  bn_add_relu_k<<<egrid, 256, 0, stream>>>(y, scale2, shift2, feat, out, nelem);
}
